// StructuredODETransition_31688268709892
// MI455X (gfx1250) — compile-verified, matrix operand order checked by reading, not run on hardware
//
#include <hip/hip_runtime.h>
#include <hip/hip_bf16.h>
#include <math.h>
#include <stdint.h>

typedef __attribute__((ext_vector_type(16))) _Float16 v16h;
typedef __attribute__((ext_vector_type(8)))  float    v8f;

#define BATCH   131072
#define NBLK    (BATCH / 32)   // 4096 blocks, 2 waves/block, 16 rows/wave

#define LDS_FENCE() asm volatile("s_wait_dscnt 0x0" ::: "memory")

// Dormand-Prince tableau (row 5 == b5; E = b5 - b4)
__constant__ float c_A6[6][6] = {
  {1.f/5, 0, 0, 0, 0, 0},
  {3.f/40, 9.f/40, 0, 0, 0, 0},
  {44.f/45, -56.f/15, 32.f/9, 0, 0, 0},
  {19372.f/6561, -25360.f/2187, 64448.f/6561, -212.f/729, 0, 0},
  {9017.f/3168, -355.f/33, 46732.f/5247, 49.f/176, -5103.f/18656, 0},
  {35.f/384, 0.f, 500.f/1113, 125.f/192, -2187.f/6784, 11.f/84}
};
__constant__ float c_E[7] = {
  35.f/384 - 5179.f/57600, 0.f, 500.f/1113 - 7571.f/16695,
  125.f/192 - 393.f/640, -2187.f/6784 + 92097.f/339200,
  11.f/84 - 187.f/2100, -1.f/40
};

__device__ __forceinline__ v8f wmma32(v16h a, v16h b, v8f c) {
  return __builtin_amdgcn_wmma_f32_16x16x32_f16(false, a, false, b, (short)0, c, false, false);
}

__device__ __forceinline__ v8f splat8(float x) {
  v8f r;
#pragma unroll
  for (int i = 0; i < 8; ++i) r[i] = x;
  return r;
}

// A-fragment (16x32 f16): lane l holds row m=l%16, K = (l/16)*8 + {0..7} and +16.
__device__ __forceinline__ v16h afrag(const _Float16* base, int rowStrideH, int m, int kOff) {
  v16h a;
  const _Float16* p = base + m * rowStrideH + kOff;
  reinterpret_cast<int4*>(&a)[0] = *reinterpret_cast<const int4*>(p);
  reinterpret_cast<int4*>(&a)[1] = *reinterpret_cast<const int4*>(p + 16);
  return a;
}

// B-fragment from fragment-packed global f16: frag*512 halves, lane*16 contiguous halves.
__device__ __forceinline__ v16h bfrag(const _Float16* pk, int frag, int l) {
  v16h b;
  const int4* p = reinterpret_cast<const int4*>(pk) + frag * 64 + l * 2;
  reinterpret_cast<int4*>(&b)[0] = p[0];
  reinterpret_cast<int4*>(&b)[1] = p[1];
  return b;
}

// ---------------- weight packing ----------------
// Pack W[OUT][IN] (row-major f32) into B-frag order: frag f = kc*NT + nt,
// per lane 16 halves: K = kc*32 + (lane/16)*8 + (hh%8) + (hh/8)*16 ; N = nt*16 + lane%16
__global__ void pack_w_kernel(const float* __restrict__ W, _Float16* __restrict__ dst,
                              int IN, int OUT, int NT, int total) {
  int idx = blockIdx.x * blockDim.x + threadIdx.x;
  if (idx >= total) return;
  int f = idx >> 9, r = idx & 511;
  int lane = r >> 4, hh = r & 15;
  int kc = f / NT, nt = f % NT;
  int K = kc * 32 + ((lane >> 4) << 3) + (hh & 7) + ((hh >> 3) << 4);
  int N = nt * 16 + (lane & 15);
  float v = (K < IN && N < OUT) ? W[N * IN + K] : 0.f;
  dst[idx] = (_Float16)v;
}

// M_B[k][n] = A[n][k] = -(L L^T)[n][k] ; K=N=64 -> 2 kc * 4 nt frags
__global__ void pack_M_kernel(const float* __restrict__ L, _Float16* __restrict__ dst) {
  int idx = blockIdx.x * blockDim.x + threadIdx.x;
  if (idx >= 8 * 512) return;
  int f = idx >> 9, r = idx & 511;
  int lane = r >> 4, hh = r & 15;
  int kc = f / 4, nt = f % 4;
  int K = kc * 32 + ((lane >> 4) << 3) + (hh & 7) + ((hh >> 3) << 4);
  int N = nt * 16 + (lane & 15);
  float s = 0.f;
  for (int t = 0; t < 64; ++t) s += L[N * 64 + t] * L[K * 64 + t];
  dst[idx] = (_Float16)(-s);
}

__global__ void init_z_kernel(const float* __restrict__ zt, float* __restrict__ Z, int n) {
  int i = blockIdx.x * blockDim.x + threadIdx.x;
  if (i < n) Z[i] = zt[i];
}

__global__ void init_ctl_kernel(float* ctl) {
  ctl[0] = 0.f;    // t
  ctl[1] = 0.1f;   // h
  ctl[2] = 0.1f;   // h_eff = min(h, 1-t)
  ctl[3] = 0.f;    // done
  ctl[4] = 0.f;    // sel (commit y->z)
}

// ---------------- one dopri5 step (all 7 stages fused) ----------------
__global__ __launch_bounds__(64)
void ode_step_kernel(float* __restrict__ Z, float* __restrict__ Y,
                     const float* __restrict__ U,
                     const _Float16* __restrict__ pkA1, const _Float16* __restrict__ pkA2,
                     const _Float16* __restrict__ pkR1, const _Float16* __restrict__ pkR2,
                     const _Float16* __restrict__ pkM,
                     const float* __restrict__ ab1, const float* __restrict__ ab2,
                     const float* __restrict__ rb1, const float* __restrict__ rb2,
                     const float* __restrict__ ctl, float* __restrict__ partials) {
  __shared__ float    kbuf[2][6][16 * 66];   // f32 k1..k6, row stride 66 (bank-friendly)
  __shared__ _Float16 zus[2][16 * 96];       // stage input [z | u | 0-pad], f16
  __shared__ _Float16 hbs[2][16 * 128];      // hidden activations, f16
  __shared__ float    red[64];

  const int tid = threadIdx.x;
  const int l   = tid & 31;
  const int wv  = tid >> 5;
  const int lm  = l & 15;
  const int lh  = l >> 4;
  const int baseRow = blockIdx.x * 32 + wv * 16;

  const float h_eff = ctl[2];
  const float sel   = ctl[4];

  _Float16* zu = zus[wv];
  _Float16* hb = hbs[wv];

  // commit previous accept and load z (C-fragment layout: m = v+8*lh, n = nt*16+lm)
  float z[4][8];
#pragma unroll
  for (int nt = 0; nt < 4; ++nt)
#pragma unroll
    for (int v = 0; v < 8; ++v) {
      int m  = v + 8 * lh;
      int gi = (baseRow + m) * 64 + nt * 16 + lm;
      float zv = Z[gi];
      if (sel > 0.5f) { zv = Y[gi]; Z[gi] = zv; }
      z[nt][v] = zv;
    }

  // stage u into zu cols 64..79, zero cols 80..95 (constant over stages)
  for (int idx = l; idx < 512; idx += 32) {
    int r = idx >> 5, c = idx & 31;
    float uv = (c < 16) ? U[(baseRow + r) * 16 + c] : 0.f;
    zu[r * 96 + 64 + c] = (_Float16)uv;
  }
  LDS_FENCE();

  float psum = 0.f;

#pragma unroll 1
  for (int st = 0; st < 7; ++st) {
    float ynew[4][8];
    // ---- build stage input into zu (stage 6 input IS y_new: FSAL) ----
#pragma unroll
    for (int nt = 0; nt < 4; ++nt)
#pragma unroll
      for (int v = 0; v < 8; ++v) {
        int m  = v + 8 * lh;
        int ci = m * 66 + nt * 16 + lm;
        float val = z[nt][v];
        if (st > 0) {
          float s = 0.f;
          for (int j = 0; j < st; ++j) s += c_A6[st - 1][j] * kbuf[wv][j][ci];
          val += h_eff * s;
        }
        if (st == 6) ynew[nt][v] = val;
        zu[m * 96 + nt * 16 + lm] = (_Float16)val;
      }
    LDS_FENCE();

    // ---- f(zi): k = zi*M + MLP_adv + MLP_res (+ biases) ----
    v8f kacc[4];
#pragma unroll
    for (int nt = 0; nt < 4; ++nt)
      kacc[nt] = splat8(ab2[nt * 16 + lm] + rb2[nt * 16 + lm]);

    for (int net = 0; net < 2; ++net) {
      const _Float16* W1 = net ? pkR1 : pkA1;
      const _Float16* W2 = net ? pkR2 : pkA2;
      const float*    b1 = net ? rb1  : ab1;

      v16h a0 = afrag(zu, 96, lm, 0  + lh * 8);
      v16h a1 = afrag(zu, 96, lm, 32 + lh * 8);
      v16h a2 = afrag(zu, 96, lm, 64 + lh * 8);
#pragma unroll
      for (int nt = 0; nt < 8; ++nt) {
        v8f acc = splat8(b1[nt * 16 + lm]);
        acc = wmma32(a0, bfrag(W1, 0 * 8 + nt, l), acc);
        acc = wmma32(a1, bfrag(W1, 1 * 8 + nt, l), acc);
        acc = wmma32(a2, bfrag(W1, 2 * 8 + nt, l), acc);
#pragma unroll
        for (int v = 0; v < 8; ++v)
          hb[(v + 8 * lh) * 128 + nt * 16 + lm] = (_Float16)tanhf(acc[v]);
      }
      LDS_FENCE();

      v16h h0 = afrag(hb, 128, lm, 0  + lh * 8);
      v16h h1 = afrag(hb, 128, lm, 32 + lh * 8);
      v16h h2 = afrag(hb, 128, lm, 64 + lh * 8);
      v16h h3 = afrag(hb, 128, lm, 96 + lh * 8);
#pragma unroll
      for (int nt = 0; nt < 4; ++nt) {
        kacc[nt] = wmma32(h0, bfrag(W2, 0 * 4 + nt, l), kacc[nt]);
        kacc[nt] = wmma32(h1, bfrag(W2, 1 * 4 + nt, l), kacc[nt]);
        kacc[nt] = wmma32(h2, bfrag(W2, 2 * 4 + nt, l), kacc[nt]);
        kacc[nt] = wmma32(h3, bfrag(W2, 3 * 4 + nt, l), kacc[nt]);
      }
    }

    // diffusion: zi (cols 0..63 of zu) * M
    {
      v16h d0 = afrag(zu, 96, lm, 0  + lh * 8);
      v16h d1 = afrag(zu, 96, lm, 32 + lh * 8);
#pragma unroll
      for (int nt = 0; nt < 4; ++nt) {
        kacc[nt] = wmma32(d0, bfrag(pkM, 0 * 4 + nt, l), kacc[nt]);
        kacc[nt] = wmma32(d1, bfrag(pkM, 1 * 4 + nt, l), kacc[nt]);
      }
    }

    if (st < 6) {
#pragma unroll
      for (int nt = 0; nt < 4; ++nt)
#pragma unroll
        for (int v = 0; v < 8; ++v)
          kbuf[wv][st][(v + 8 * lh) * 66 + nt * 16 + lm] = kacc[nt][v];
      LDS_FENCE();
    } else {
      // k7 -> error only; y_new already in ynew
#pragma unroll
      for (int nt = 0; nt < 4; ++nt)
#pragma unroll
        for (int v = 0; v < 8; ++v) {
          int m  = v + 8 * lh;
          int ci = m * 66 + nt * 16 + lm;
          float e = c_E[6] * kacc[nt][v];
          for (int j = 0; j < 6; ++j) e += c_E[j] * kbuf[wv][j][ci];
          e *= h_eff;
          float yv = ynew[nt][v];
          float sc = 1e-3f + 1e-3f * fmaxf(fabsf(z[nt][v]), fabsf(yv));
          float q  = e / sc;
          psum += q * q;
          Y[(baseRow + m) * 64 + nt * 16 + lm] = yv;
        }
    }
  }

  red[tid] = psum;
  __syncthreads();
  if (tid == 0) {
    float s = 0.f;
    for (int i = 0; i < 64; ++i) s += red[i];
    partials[blockIdx.x] = s;
  }
}

// ---------------- step-size / accept control (deterministic reduction) ----------------
__global__ __launch_bounds__(256)
void ode_ctrl_kernel(const float* __restrict__ partials, float* __restrict__ ctl) {
  __shared__ float red[256];
  int tid = threadIdx.x;
  float s = 0.f;
  for (int i = tid; i < NBLK; i += 256) s += partials[i];
  red[tid] = s;
  __syncthreads();
  for (int off = 128; off > 0; off >>= 1) {
    if (tid < off) red[tid] += red[tid + off];
    __syncthreads();
  }
  if (tid == 0) {
    float t = ctl[0], h = ctl[1], heff = ctl[2];
    bool done = ctl[3] > 0.5f;
    float en = sqrtf(red[0] / (float)(BATCH * 64));
    en = fmaxf(en, 1e-8f);
    bool accept = en <= 1.0f;
    float factor = fminf(fmaxf(0.9f * powf(en, -0.2f), 0.2f), 10.0f);
    float hn = done ? h : heff * factor;
    float tn = done ? t : (accept ? t + heff : t);
    ctl[0] = tn;
    ctl[1] = hn;
    ctl[2] = fminf(hn, 1.0f - tn);
    ctl[3] = (tn >= 1.0f - 1e-7f) ? 1.0f : 0.0f;
    ctl[4] = (accept && !done) ? 1.0f : 0.0f;
  }
}

// ---------------- observation head: zt1 + relu MLP ----------------
__global__ __launch_bounds__(64)
void ode_head_kernel(const float* __restrict__ Z, float* __restrict__ Yz, float* __restrict__ Yy,
                     const float* __restrict__ U,
                     const _Float16* __restrict__ pkO1, const _Float16* __restrict__ pkO2,
                     const float* __restrict__ ob1, const float* __restrict__ ob2,
                     const float* __restrict__ ctl) {
  __shared__ _Float16 zus[2][16 * 96];
  __shared__ _Float16 hbs[2][16 * 128];

  const int tid = threadIdx.x;
  const int l   = tid & 31;
  const int wv  = tid >> 5;
  const int lm  = l & 15;
  const int lh  = l >> 4;
  const int baseRow = blockIdx.x * 32 + wv * 16;

  const float sel = ctl[4];
  _Float16* zu = zus[wv];
  _Float16* hb = hbs[wv];

#pragma unroll
  for (int nt = 0; nt < 4; ++nt)
#pragma unroll
    for (int v = 0; v < 8; ++v) {
      int m  = v + 8 * lh;
      int gi = (baseRow + m) * 64 + nt * 16 + lm;
      float zv = Z[gi];
      if (sel > 0.5f) zv = Yz[gi];
      Yz[gi] = zv;                          // final zt1
      zu[m * 96 + nt * 16 + lm] = (_Float16)zv;
    }
  for (int idx = l; idx < 512; idx += 32) {
    int r = idx >> 5, c = idx & 31;
    float uv = (c < 16) ? U[(baseRow + r) * 16 + c] : 0.f;
    zu[r * 96 + 64 + c] = (_Float16)uv;
  }
  LDS_FENCE();

  v16h a0 = afrag(zu, 96, lm, 0  + lh * 8);
  v16h a1 = afrag(zu, 96, lm, 32 + lh * 8);
  v16h a2 = afrag(zu, 96, lm, 64 + lh * 8);
#pragma unroll
  for (int nt = 0; nt < 8; ++nt) {
    v8f acc = splat8(ob1[nt * 16 + lm]);
    acc = wmma32(a0, bfrag(pkO1, 0 * 8 + nt, l), acc);
    acc = wmma32(a1, bfrag(pkO1, 1 * 8 + nt, l), acc);
    acc = wmma32(a2, bfrag(pkO1, 2 * 8 + nt, l), acc);
#pragma unroll
    for (int v = 0; v < 8; ++v)
      hb[(v + 8 * lh) * 128 + nt * 16 + lm] = (_Float16)fmaxf(acc[v], 0.f);
  }
  LDS_FENCE();

  v16h h0 = afrag(hb, 128, lm, 0  + lh * 8);
  v16h h1 = afrag(hb, 128, lm, 32 + lh * 8);
  v16h h2 = afrag(hb, 128, lm, 64 + lh * 8);
  v16h h3 = afrag(hb, 128, lm, 96 + lh * 8);
#pragma unroll
  for (int nt = 0; nt < 2; ++nt) {
    int n = nt * 16 + lm;
    v8f acc = splat8(n < 20 ? ob2[n] : 0.f);
    acc = wmma32(h0, bfrag(pkO2, 0 * 2 + nt, l), acc);
    acc = wmma32(h1, bfrag(pkO2, 1 * 2 + nt, l), acc);
    acc = wmma32(h2, bfrag(pkO2, 2 * 2 + nt, l), acc);
    acc = wmma32(h3, bfrag(pkO2, 3 * 2 + nt, l), acc);
    if (n < 20) {
#pragma unroll
      for (int v = 0; v < 8; ++v)
        Yy[(baseRow + v + 8 * lh) * 20 + n] = acc[v];
    }
  }
}

extern "C" void kernel_launch(void* const* d_in, const int* in_sizes, int n_in,
                              void* d_out, int out_size, void* d_ws, size_t ws_size,
                              hipStream_t stream) {
  (void)in_sizes; (void)n_in; (void)out_size; (void)ws_size;
  const float* zt  = (const float*)d_in[0];
  /* d_in[1] = dt (unused by reference) */
  const float* ut  = (const float*)d_in[2];
  const float* L   = (const float*)d_in[3];
  const float* aw1 = (const float*)d_in[4];
  const float* ab1 = (const float*)d_in[5];
  const float* aw2 = (const float*)d_in[6];
  const float* ab2 = (const float*)d_in[7];
  const float* rw1 = (const float*)d_in[8];
  const float* rb1 = (const float*)d_in[9];
  const float* rw2 = (const float*)d_in[10];
  const float* rb2 = (const float*)d_in[11];
  const float* ow1 = (const float*)d_in[12];
  const float* ob1 = (const float*)d_in[13];
  const float* ow2 = (const float*)d_in[14];
  const float* ob2 = (const float*)d_in[15];

  float* Z        = (float*)d_ws;                       // B*64 f32 state
  float* partials = Z + (size_t)BATCH * 64;             // NBLK f32
  float* ctl      = partials + NBLK;                    // 8 f32
  _Float16* pk    = (_Float16*)(ctl + 8);               // 16B-aligned by construction
  _Float16* pkA1 = pk;                 // 24 frags
  _Float16* pkA2 = pkA1 + 24 * 512;    // 16 frags
  _Float16* pkR1 = pkA2 + 16 * 512;    // 24 frags
  _Float16* pkR2 = pkR1 + 24 * 512;    // 16 frags
  _Float16* pkM  = pkR2 + 16 * 512;    //  8 frags
  _Float16* pkO1 = pkM  + 8 * 512;     // 24 frags
  _Float16* pkO2 = pkO1 + 24 * 512;    //  8 frags

  float* Yz = (float*)d_out;                  // zt1 region doubles as y_new buffer
  float* Yy = Yz + (size_t)BATCH * 64;        // yt1 region

  init_z_kernel<<<(BATCH * 64 + 255) / 256, 256, 0, stream>>>(zt, Z, BATCH * 64);
  init_ctl_kernel<<<1, 1, 0, stream>>>(ctl);
  pack_w_kernel<<<(24 * 512 + 255) / 256, 256, 0, stream>>>(aw1, pkA1, 80, 128, 8, 24 * 512);
  pack_w_kernel<<<(16 * 512 + 255) / 256, 256, 0, stream>>>(aw2, pkA2, 128, 64, 4, 16 * 512);
  pack_w_kernel<<<(24 * 512 + 255) / 256, 256, 0, stream>>>(rw1, pkR1, 80, 128, 8, 24 * 512);
  pack_w_kernel<<<(16 * 512 + 255) / 256, 256, 0, stream>>>(rw2, pkR2, 128, 64, 4, 16 * 512);
  pack_M_kernel<<<(8 * 512 + 255) / 256, 256, 0, stream>>>(L, pkM);
  pack_w_kernel<<<(24 * 512 + 255) / 256, 256, 0, stream>>>(ow1, pkO1, 80, 128, 8, 24 * 512);
  pack_w_kernel<<<(8 * 512 + 255) / 256, 256, 0, stream>>>(ow2, pkO2, 128, 20, 2, 8 * 512);

  for (int s = 0; s < 40; ++s) {
    ode_step_kernel<<<NBLK, 64, 0, stream>>>(Z, Yz, ut, pkA1, pkA2, pkR1, pkR2, pkM,
                                             ab1, ab2, rb1, rb2, ctl, partials);
    ode_ctrl_kernel<<<1, 256, 0, stream>>>(partials, ctl);
  }
  ode_head_kernel<<<NBLK, 64, 0, stream>>>(Z, Yz, Yy, ut, pkO1, pkO2, ob1, ob2, ctl);
}
